// _GlobalWorkspace_85057532330646
// MI455X (gfx1250) — matrix operand order repaired and checked by reading, not compiled, not run on hardware
//
#include <hip/hip_runtime.h>
#include <hip/hip_bf16.h>

// GlobalWorkspace attention, restructured for MI455X (gfx1250, wave32, WMMA).
//   P   = (workspace @ Wq_w^T) @ Wk_w          (4x512, f32 precompute)
//   MqT so that T = hidden @ (Wq_r^T Wk_r)
//   Mvo = Wo @ Wv_r
// Big GEMMs (bf16 in, f32 accum, v_wmma_f32_16x16x32_bf16, software-pipelined):
//   Vw  = hidden_bf16 @ Wv_w^T        [65536 x 512]
//   T   = hidden_bf16 @ Mq            [65536 x 512]
//   VrO = ws_upd_bf16 @ Mvo^T         [32768 x 512]  (f32 out)
// Per-batch softmax/LayerNorm: warp-per-row, __shfl_xor wave32 reductions.

#define D_DIM   512
#define K_MOD   8
#define S_SLOT  4
#define B_BATCH 8192
#define SCALE_F 0.04419417382415922f   // 512^-0.5
#define LN_EPS_F 1e-5f

typedef __bf16 bf16_t;
typedef __attribute__((ext_vector_type(8)))  __bf16 v8bf;
typedef __attribute__((ext_vector_type(16))) __bf16 v16bf;
typedef __attribute__((ext_vector_type(8)))  float  v8f;

union FragU { v16bf v; v8bf h[2]; };

// A-matrix 16x32 bf16 fragment (ISA 7.12.2): lane l, m = l&15, half = l>>4,
// VGPR0-3 hold K = half*8 .. +7, VGPR4-7 hold K = 16 + half*8 .. +7.
__device__ __forceinline__ v16bf load_frag_A(const bf16_t* base, int row0, int ld,
                                             int kbase, int lane) {
  int r = lane & 15, hh = lane >> 4;
  const bf16_t* p = base + (size_t)(row0 + r) * ld + kbase + hh * 8;
  FragU f;
  f.h[0] = *(const v8bf*)p;         // K = kbase + hh*8  .. +7
  f.h[1] = *(const v8bf*)(p + 16);  // K = kbase + 16 + hh*8 .. +7
  return f.v;
}

// B-matrix 32x16 bf16 fragment: lane l, n = l&15, half = l>>4,
// lane holds K = half*8 .. +7 and 16 + half*8 .. +7. B stored row-major as Bm[n][k].
__device__ __forceinline__ v16bf load_frag_B(const bf16_t* base, int col0, int ld,
                                             int kbase, int lane) {
  int n = lane & 15, hh = lane >> 4;
  const bf16_t* p = base + (size_t)(col0 + n) * ld + kbase + hh * 8;
  FragU f;
  f.h[0] = *(const v8bf*)p;         // K = kbase + hh*8  .. +7
  f.h[1] = *(const v8bf*)(p + 16);  // K = kbase + 16 + hh*8 .. +7
  return f.v;
}

__device__ __forceinline__ v8f wmma_bf16(v16bf a, v16bf b, v8f c) {
  return __builtin_amdgcn_wmma_f32_16x16x32_bf16(false, a, false, b,
                                                 (short)0, c, false, false);
}

// C = A[M,Kd] * Bm[N,Kd]^T.  Wave tile 32x64 (8 WMMAs / 12 b128 loads per
// k-step), block = 8 waves stacked in M -> 256x64.  K loop software-pipelined:
// next k-step's fragments are loaded before the current step's WMMAs so loads
// overlap matrix math instead of draining loadcnt before every WMMA.
template<bool OUT_BF16, int Kd>
__global__ __launch_bounds__(256) void gemm_bf16_nt(const bf16_t* __restrict__ A,
                                                    const bf16_t* __restrict__ Bm,
                                                    void* __restrict__ Cv,
                                                    int N) {
  const int lane  = threadIdx.x & 31;
  const int wave  = threadIdx.x >> 5;
  const int mbase = blockIdx.x * 256 + wave * 32;
  const int nbase = blockIdx.y * 64;

  v8f acc0[4] = {};
  v8f acc1[4] = {};

  // prologue: k = 0 fragments
  v16bf a0 = load_frag_A(A, mbase,      Kd, 0, lane);
  v16bf a1 = load_frag_A(A, mbase + 16, Kd, 0, lane);
  v16bf b0 = load_frag_B(Bm, nbase,      Kd, 0, lane);
  v16bf b1 = load_frag_B(Bm, nbase + 16, Kd, 0, lane);
  v16bf b2 = load_frag_B(Bm, nbase + 32, Kd, 0, lane);
  v16bf b3 = load_frag_B(Bm, nbase + 48, Kd, 0, lane);

#pragma unroll 2
  for (int k = 32; k < Kd; k += 32) {
    // prefetch next k-step while current WMMAs run
    v16bf na0 = load_frag_A(A, mbase,      Kd, k, lane);
    v16bf na1 = load_frag_A(A, mbase + 16, Kd, k, lane);
    v16bf nb0 = load_frag_B(Bm, nbase,      Kd, k, lane);
    v16bf nb1 = load_frag_B(Bm, nbase + 16, Kd, k, lane);
    v16bf nb2 = load_frag_B(Bm, nbase + 32, Kd, k, lane);
    v16bf nb3 = load_frag_B(Bm, nbase + 48, Kd, k, lane);

    acc0[0] = wmma_bf16(a0, b0, acc0[0]);
    acc1[0] = wmma_bf16(a1, b0, acc1[0]);
    acc0[1] = wmma_bf16(a0, b1, acc0[1]);
    acc1[1] = wmma_bf16(a1, b1, acc1[1]);
    acc0[2] = wmma_bf16(a0, b2, acc0[2]);
    acc1[2] = wmma_bf16(a1, b2, acc1[2]);
    acc0[3] = wmma_bf16(a0, b3, acc0[3]);
    acc1[3] = wmma_bf16(a1, b3, acc1[3]);

    a0 = na0; a1 = na1; b0 = nb0; b1 = nb1; b2 = nb2; b3 = nb3;
  }
  // epilogue: last k-step
  acc0[0] = wmma_bf16(a0, b0, acc0[0]);
  acc1[0] = wmma_bf16(a1, b0, acc1[0]);
  acc0[1] = wmma_bf16(a0, b1, acc0[1]);
  acc1[1] = wmma_bf16(a1, b1, acc1[1]);
  acc0[2] = wmma_bf16(a0, b2, acc0[2]);
  acc1[2] = wmma_bf16(a1, b2, acc1[2]);
  acc0[3] = wmma_bf16(a0, b3, acc0[3]);
  acc1[3] = wmma_bf16(a1, b3, acc1[3]);

  // C/D layout: lane l -> N = l&15, VGPR r -> M = r + 8*(l>>4)
  const int hh = lane >> 4, nc = lane & 15;
#pragma unroll
  for (int nt = 0; nt < 4; ++nt) {
#pragma unroll
    for (int r = 0; r < 8; ++r) {
      size_t i0 = (size_t)(mbase + r + 8 * hh) * N + (size_t)(nbase + nt * 16 + nc);
      size_t i1 = (size_t)(mbase + 16 + r + 8 * hh) * N + (size_t)(nbase + nt * 16 + nc);
      float v0 = acc0[nt][r];
      float v1 = acc1[nt][r];
      if (OUT_BF16) {
        ((bf16_t*)Cv)[i0] = (bf16_t)v0;
        ((bf16_t*)Cv)[i1] = (bf16_t)v1;
      } else {
        ((float*)Cv)[i0] = v0;
        ((float*)Cv)[i1] = v1;
      }
    }
  }
}

__global__ __launch_bounds__(256) void cvt_bf16(const float* __restrict__ src,
                                                bf16_t* __restrict__ dst, long long n) {
  long long i = ((long long)blockIdx.x * blockDim.x + threadIdx.x) * 8;
  if (i + 8 <= n) {
#pragma unroll
    for (int j = 0; j < 8; ++j) dst[i + j] = (bf16_t)src[i + j];
  }
}

// 512x512 transpose + f32->bf16 (dst[c][r] = src[r][c])
__global__ __launch_bounds__(256) void cvt_bf16_T512(const float* __restrict__ src,
                                                     bf16_t* __restrict__ dst) {
  int i = blockIdx.x * 256 + threadIdx.x;     // 262144 threads
  int r = i >> 9, c = i & 511;
  dst[(size_t)c * 512 + r] = (bf16_t)src[(size_t)r * 512 + c];
}

// Qw[s,e] = sum_d workspace[s,d] * Wq_w[e,d]   (4x512)
__global__ __launch_bounds__(256) void qw_kernel(const float* __restrict__ wsl,
                                                 const float* __restrict__ Wq_w,
                                                 float* __restrict__ Qw) {
  int i = blockIdx.x * 256 + threadIdx.x;     // 2048 threads
  int s = i >> 9, e = i & 511;
  float acc = 0.f;
  for (int d = 0; d < 512; ++d) acc += wsl[s * 512 + d] * Wq_w[e * 512 + d];
  Qw[i] = acc;
}

// P[s,d] = sum_e Qw[s,e] * Wk_w[e,d]   (4x512)
__global__ __launch_bounds__(256) void p_kernel(const float* __restrict__ Qw,
                                                const float* __restrict__ Wk_w,
                                                float* __restrict__ P) {
  int i = blockIdx.x * 256 + threadIdx.x;
  int s = i >> 9, d = i & 511;
  float acc = 0.f;
  for (int e = 0; e < 512; ++e) acc += Qw[s * 512 + e] * Wk_w[e * 512 + d];
  P[i] = acc;
}

// Write attention + LN. One block per batch; warp s handles slot s.
__global__ __launch_bounds__(128) void phase_write(const float* __restrict__ hidden,
    const int* __restrict__ mask, const float* __restrict__ wsl,
    const float* __restrict__ P, const bf16_t* __restrict__ Vw,
    const float* __restrict__ g_ws, const float* __restrict__ b_ws,
    bf16_t* __restrict__ WSU) {
  const int b = blockIdx.x;
  const int s = threadIdx.x >> 5;
  const int lane = threadIdx.x & 31;
  const float* hb = hidden + (size_t)b * K_MOD * D_DIM;

  float attn[K_MOD];
  float mx = -__builtin_inff();
  for (int k = 0; k < K_MOD; ++k) {
    float part = 0.f;
#pragma unroll
    for (int j = 0; j < 16; ++j) {
      int d = lane + 32 * j;
      part += hb[k * D_DIM + d] * P[s * D_DIM + d];
    }
#pragma unroll
    for (int o = 16; o > 0; o >>= 1) part += __shfl_xor(part, o, 32);
    float sc = (mask[b * K_MOD + k] != 0) ? part * SCALE_F : -__builtin_inff();
    attn[k] = sc;
    mx = fmaxf(mx, sc);
  }
  if (mx == -__builtin_inff()) {            // all masked -> nan_to_num -> 0
    for (int k = 0; k < K_MOD; ++k) attn[k] = 0.f;
  } else {
    float ssum = 0.f;
    for (int k = 0; k < K_MOD; ++k) { attn[k] = __expf(attn[k] - mx); ssum += attn[k]; }
    float inv = 1.f / ssum;
    for (int k = 0; k < K_MOD; ++k) attn[k] *= inv;
  }

  const bf16_t* Vb = Vw + (size_t)b * K_MOD * D_DIM;
  float acc[16], sum = 0.f, sq = 0.f;
#pragma unroll
  for (int j = 0; j < 16; ++j) {
    int d = lane + 32 * j;
    float v = wsl[s * D_DIM + d];
#pragma unroll
    for (int k = 0; k < K_MOD; ++k) v += attn[k] * (float)Vb[k * D_DIM + d];
    acc[j] = v; sum += v; sq += v * v;
  }
#pragma unroll
  for (int o = 16; o > 0; o >>= 1) { sum += __shfl_xor(sum, o, 32); sq += __shfl_xor(sq, o, 32); }
  float mean = sum * (1.f / D_DIM);
  float var  = sq * (1.f / D_DIM) - mean * mean;
  float rinv = rsqrtf(var + LN_EPS_F);
  bf16_t* Wr = WSU + ((size_t)b * S_SLOT + s) * D_DIM;
#pragma unroll
  for (int j = 0; j < 16; ++j) {
    int d = lane + 32 * j;
    Wr[d] = (bf16_t)((acc[j] - mean) * rinv * g_ws[d] + b_ws[d]);
  }
}

// Read attention + out-proj (pre-folded) + LN. One block per batch; warp k = module k.
__global__ __launch_bounds__(256) void phase_read(const float* __restrict__ hidden,
    const bf16_t* __restrict__ T, const bf16_t* __restrict__ WSU,
    const float* __restrict__ VrO, const float* __restrict__ g_out,
    const float* __restrict__ b_out, float* __restrict__ out) {
  const int b = blockIdx.x;
  const int k = threadIdx.x >> 5;
  const int lane = threadIdx.x & 31;

  const bf16_t* Tk = T + ((size_t)b * K_MOD + k) * D_DIM;
  float t[16];
#pragma unroll
  for (int j = 0; j < 16; ++j) t[j] = (float)Tk[lane + 32 * j];

  float attn[S_SLOT];
  float mx = -__builtin_inff();
  for (int s = 0; s < S_SLOT; ++s) {
    const bf16_t* Wr = WSU + ((size_t)b * S_SLOT + s) * D_DIM;
    float part = 0.f;
#pragma unroll
    for (int j = 0; j < 16; ++j) part += t[j] * (float)Wr[lane + 32 * j];
#pragma unroll
    for (int o = 16; o > 0; o >>= 1) part += __shfl_xor(part, o, 32);
    attn[s] = part * SCALE_F;
    mx = fmaxf(mx, attn[s]);
  }
  float ssum = 0.f;
  for (int s = 0; s < S_SLOT; ++s) { attn[s] = __expf(attn[s] - mx); ssum += attn[s]; }
  float inv = 1.f / ssum;
  for (int s = 0; s < S_SLOT; ++s) attn[s] *= inv;

  const float* Vb = VrO + (size_t)b * S_SLOT * D_DIM;
  const float* hk = hidden + ((size_t)b * K_MOD + k) * D_DIM;
  float acc[16], sum = 0.f, sq = 0.f;
#pragma unroll
  for (int j = 0; j < 16; ++j) {
    int d = lane + 32 * j;
    float v = hk[d];
#pragma unroll
    for (int s = 0; s < S_SLOT; ++s) v += attn[s] * Vb[s * D_DIM + d];
    acc[j] = v; sum += v; sq += v * v;
  }
#pragma unroll
  for (int o = 16; o > 0; o >>= 1) { sum += __shfl_xor(sum, o, 32); sq += __shfl_xor(sq, o, 32); }
  float mean = sum * (1.f / D_DIM);
  float var  = sq * (1.f / D_DIM) - mean * mean;
  float rinv = rsqrtf(var + LN_EPS_F);
  float* orow = out + ((size_t)b * K_MOD + k) * D_DIM;
#pragma unroll
  for (int j = 0; j < 16; ++j) {
    int d = lane + 32 * j;
    orow[d] = (acc[j] - mean) * rinv * g_out[d] + b_out[d];
  }
}

extern "C" void kernel_launch(void* const* d_in, const int* in_sizes, int n_in,
                              void* d_out, int out_size, void* d_ws, size_t ws_size,
                              hipStream_t stream) {
  const float* hidden   = (const float*)d_in[0];
  const int*   mask     = (const int*)  d_in[1];
  const float* wsl      = (const float*)d_in[2];
  const float* Wq_w     = (const float*)d_in[3];
  const float* Wk_w     = (const float*)d_in[4];
  const float* Wv_w     = (const float*)d_in[5];
  const float* Wq_r     = (const float*)d_in[6];
  const float* Wk_r     = (const float*)d_in[7];
  const float* Wv_r     = (const float*)d_in[8];
  const float* Wo       = (const float*)d_in[9];
  const float* g_ws     = (const float*)d_in[10];
  const float* b_ws     = (const float*)d_in[11];
  const float* g_out    = (const float*)d_in[12];
  const float* b_out    = (const float*)d_in[13];
  float* out = (float*)d_out;

  const size_t MK = (size_t)B_BATCH * K_MOD;   // 65536
  const size_t MS = (size_t)B_BATCH * S_SLOT;  // 32768

  size_t off = 0;
  auto alloc = [&](size_t bytes) -> void* {
    void* p = (char*)d_ws + off;
    off += (bytes + 255) & ~(size_t)255;
    return p;
  };
  bf16_t* Ah    = (bf16_t*)alloc(MK * D_DIM * sizeof(bf16_t));       // hidden bf16
  bf16_t* Vw    = (bf16_t*)alloc(MK * D_DIM * sizeof(bf16_t));
  bf16_t* Tb    = (bf16_t*)alloc(MK * D_DIM * sizeof(bf16_t));
  bf16_t* WSU   = (bf16_t*)alloc(MS * D_DIM * sizeof(bf16_t));       // ws_upd bf16
  float*  VrO   = (float*) alloc(MS * D_DIM * sizeof(float));
  bf16_t* Wv_wb = (bf16_t*)alloc(512 * 512 * sizeof(bf16_t));
  bf16_t* Wob   = (bf16_t*)alloc(512 * 512 * sizeof(bf16_t));
  bf16_t* WqrT  = (bf16_t*)alloc(512 * 512 * sizeof(bf16_t));
  bf16_t* WkrT  = (bf16_t*)alloc(512 * 512 * sizeof(bf16_t));
  bf16_t* WvrT  = (bf16_t*)alloc(512 * 512 * sizeof(bf16_t));
  bf16_t* MqT   = (bf16_t*)alloc(512 * 512 * sizeof(bf16_t));
  bf16_t* Mvo   = (bf16_t*)alloc(512 * 512 * sizeof(bf16_t));
  float*  Qw    = (float*) alloc(S_SLOT * D_DIM * sizeof(float));
  float*  P     = (float*) alloc(S_SLOT * D_DIM * sizeof(float));
  (void)ws_size; (void)in_sizes; (void)n_in; (void)out_size;

  // --- conversions / tiny precomputes ---
  cvt_bf16<<<16384, 256, 0, stream>>>(hidden, Ah, (long long)MK * D_DIM);
  cvt_bf16<<<128, 256, 0, stream>>>(Wv_w, Wv_wb, 512LL * 512);
  cvt_bf16<<<128, 256, 0, stream>>>(Wo,   Wob,   512LL * 512);
  cvt_bf16_T512<<<1024, 256, 0, stream>>>(Wq_r, WqrT);
  cvt_bf16_T512<<<1024, 256, 0, stream>>>(Wk_r, WkrT);
  cvt_bf16_T512<<<1024, 256, 0, stream>>>(Wv_r, WvrT);
  qw_kernel<<<8, 256, 0, stream>>>(wsl, Wq_w, Qw);
  p_kernel<<<8, 256, 0, stream>>>(Qw, Wk_w, P);

  // --- folded weight products (WMMA, 512x512x512) ---
  dim3 gs(2, 8);                                                       // 512/256, 512/64
  gemm_bf16_nt<true, 512><<<gs, 256, 0, stream>>>(WkrT, WqrT, MqT, 512); // MqT[d'][d]
  gemm_bf16_nt<true, 512><<<gs, 256, 0, stream>>>(Wob,  WvrT, Mvo, 512); // Mvo[f][d]

  // --- big WMMA GEMMs ---
  dim3 g1(256, 8);                                                     // 65536/256, 512/64
  gemm_bf16_nt<true, 512><<<g1, 256, 0, stream>>>(Ah, Wv_wb, Vw, 512);
  gemm_bf16_nt<true, 512><<<g1, 256, 0, stream>>>(Ah, MqT,   Tb, 512);

  // --- write attention + LN -> ws_upd ---
  phase_write<<<B_BATCH, 128, 0, stream>>>(hidden, mask, wsl, P, Vw, g_ws, b_ws, WSU);

  // --- VrO = ws_upd @ Mvo^T (f32 out) ---
  dim3 g3(128, 8);                                                     // 32768/256
  gemm_bf16_nt<false, 512><<<g3, 256, 0, stream>>>(WSU, Mvo, VrO, 512);

  // --- read attention + folded out-proj + LN -> out ---
  phase_read<<<B_BATCH, 256, 0, stream>>>(hidden, Tb, WSU, VrO, g_out, b_out, out);
}
